// DotAttention_5334349381889
// MI455X (gfx1250) — matrix operand order repaired and checked by reading, not compiled, not run on hardware
//
#include <hip/hip_runtime.h>
#include <hip/hip_bf16.h>
#include <stdint.h>

// Problem constants from the reference
#define BB   8
#define LL   2048
#define DIN  256
#define DATT 128

// Padded LDS row stride for the 16 x 2048 f16 score/probability tile.
// Dword stride = 2058/2 = 1029 = 5 (mod 64 banks) -> conflict-free WMMA A loads.
#define SP   2058

typedef __attribute__((ext_vector_type(16))) _Float16 v16h;
typedef __attribute__((ext_vector_type(8)))  float    v8f;

union HalfPack { uint32_t u[8]; v16h v; };

// ---------------------------------------------------------------------------
// Kernel 1: projections + ELU, write f16 inp/mem.
// One block per (b,l) row; 128 threads, each owns one of the 128 attn dims.
// ---------------------------------------------------------------------------
__global__ __launch_bounds__(128)
void proj_kernel(const float* __restrict__ ctx, const float* __restrict__ qry,
                 const float* __restrict__ Win, const float* __restrict__ Wmem,
                 _Float16* __restrict__ inp16, _Float16* __restrict__ mem16)
{
    __shared__ float xc[DIN];
    __shared__ float xq[DIN];
    const int row = blockIdx.x;           // b*L + l
    const int t   = threadIdx.x;          // 0..127 -> output feature
    const float* c = ctx + (size_t)row * DIN;
    const float* q = qry + (size_t)row * DIN;
    xc[t]       = c[t];
    xc[t + 128] = c[t + 128];
    xq[t]       = q[t];
    xq[t + 128] = q[t + 128];
    __syncthreads();

    float a0 = 0.f, a1 = 0.f;
    #pragma unroll 8
    for (int k = 0; k < DIN; ++k) {
        a0 = fmaf(xc[k], Win [k * DATT + t], a0);   // coalesced over t
        a1 = fmaf(xq[k], Wmem[k * DATT + t], a1);
    }
    // ELU, alpha = 0.01
    a0 = (a0 > 0.f) ? a0 : 0.01f * (__expf(a0) - 1.f);
    a1 = (a1 > 0.f) ? a1 : 0.01f * (__expf(a1) - 1.f);
    inp16[(size_t)row * DATT + t] = (_Float16)a0;
    mem16[(size_t)row * DATT + t] = (_Float16)a1;
}

// ---------------------------------------------------------------------------
// Kernel 2: query [B,L,DIN] f32 -> queryT [B,DIN,L] f16 (coalesced writes).
// ---------------------------------------------------------------------------
__global__ __launch_bounds__(256)
void transposeq_kernel(const float* __restrict__ qry, _Float16* __restrict__ qT)
{
    const int b = blockIdx.x >> 8;
    const int d = blockIdx.x & 255;
    for (int l = threadIdx.x; l < LL; l += 256)
        qT[((size_t)b * DIN + d) * LL + l] =
            (_Float16)qry[((size_t)b * LL + l) * DIN + d];
}

// ---------------------------------------------------------------------------
// Kernel 3: fused attention. One workgroup (4 waves) per 16-row c-tile.
// LDS: Sf16[16][SP] (64.3KB, padded, in-place exp) + rowinv[16] floats.
// ~64.4KB total -> up to 4 workgroups (16 waves) per WGP for latency hiding.
// ---------------------------------------------------------------------------
__global__ __launch_bounds__(128)
void attn_kernel(const _Float16* __restrict__ inp16,
                 const _Float16* __restrict__ mem16,
                 const _Float16* __restrict__ qT,
                 const int* __restrict__ qmask,
                 float* __restrict__ out)
{
    extern __shared__ char smem[];
    _Float16* S16    = (_Float16*)smem;                       // [16][SP]
    uint32_t* Su     = (uint32_t*)smem;
    float*    rowinv = (float*)(smem + (size_t)16 * SP * 2);  // [16]

    const int tid  = threadIdx.x;
    const int wave = tid >> 5;
    const int lane = tid & 31;
    const int hh   = lane >> 4;     // lane half (0/1)
    const int n16  = lane & 15;     // lane-within-half
    const int b    = blockIdx.x >> 7;        // L/16 = 128 tiles per batch
    const int c0   = (blockIdx.x & 127) << 4;

    const float scale = 0.08838834764831845f;  // 1/sqrt(128)
    // f16-representable "minus infinity": keeps all-masked rows finite so the
    // softmax degenerates to uniform (matching the reference's -FLT_MAX path).
    const _Float16 HMIN = (_Float16)(-65504.0f);

    // ---- Preload A-tile (inp rows c0..c0+15, K=0..127) in WMMA A layout ----
    // lane: m = n16; VGPR j: k = 2*(j&3) + 16*(j>=4) + 8*hh  (pair k,k+1)
    uint32_t areg[32];
    {
        const uint32_t* abase =
            (const uint32_t*)(inp16 + (size_t)(b * LL + c0 + n16) * DATT);
        #pragma unroll
        for (int kt = 0; kt < 4; ++kt)
            #pragma unroll
            for (int j = 0; j < 8; ++j) {
                const int k = 32 * kt + 2 * (j & 3) + ((j & 4) ? 16 : 0) + hh * 8;
                areg[kt * 8 + j] = abase[k >> 1];
            }
    }
    const int* qm = qmask + b * LL;

    // ---- Stage 1: S = inp @ mem^T * scale, masked, stored f16 in LDS ----
    for (int qt = wave; qt < LL / 16; qt += 4) {
        const int q0 = qt * 16;
        v8f acc = {};
        const uint32_t* bbase =
            (const uint32_t*)(mem16 + (size_t)(b * LL + q0 + n16) * DATT);
        #pragma unroll
        for (int kt = 0; kt < 4; ++kt) {
            HalfPack aU, bU;
            #pragma unroll
            for (int j = 0; j < 8; ++j) {
                aU.u[j] = areg[kt * 8 + j];
                // B layout: lane n = n16; VGPR j -> k-pair (2*(j&3) + 16*(j>=4) + 8*hh, +1)
                const int k = 32 * kt + 2 * (j & 3) + ((j & 4) ? 16 : 0) + hh * 8;
                bU.u[j] = bbase[k >> 1];
            }
            acc = __builtin_amdgcn_wmma_f32_16x16x32_f16(
                false, aU.v, false, bU.v, (short)0, acc, false, false);
        }
        const int  q     = q0 + n16;
        const bool valid = qm[q] > 0;
        #pragma unroll
        for (int r = 0; r < 8; ++r) {
            const int m = r + hh * 8;          // C/D layout: VGPR r -> m
            _Float16 v = (_Float16)(acc[r] * scale);
            if (!valid || (q == c0 + m)) v = HMIN;
            S16[m * SP + q] = v;
        }
    }
    __syncthreads();

    // ---- Stage 2: softmax over q per row (4 rows/wave); exp in place,
    //      1/sum deferred to the output epilogue via rowinv[]. ----
    for (int m = wave * 4; m < wave * 4 + 4; ++m) {
        _Float16* row = S16 + m * SP;
        float mx = -INFINITY;
        for (int q = lane; q < LL; q += 32) mx = fmaxf(mx, (float)row[q]);
        #pragma unroll
        for (int off = 16; off; off >>= 1) mx = fmaxf(mx, __shfl_xor(mx, off, 32));
        float sum = 0.f;
        for (int q = lane; q < LL; q += 32) {
            const float e = __expf((float)row[q] - mx);
            row[q] = (_Float16)e;
            sum += e;
        }
        #pragma unroll
        for (int off = 16; off; off >>= 1) sum += __shfl_xor(sum, off, 32);
        if (lane == 0) rowinv[m] = 1.0f / sum;
    }
    __syncthreads();

    // ---- Stage 3: O = P @ query  (M=16, K=2048, N=256) ----
    // Each wave owns N-tiles {wave, wave+4, wave+8, wave+12}.
    v8f acc[4];
    #pragma unroll
    for (int t = 0; t < 4; ++t) acc[t] = (v8f){};

    const uint32_t* qb[4];
    #pragma unroll
    for (int t = 0; t < 4; ++t) {
        const int col = (wave + 4 * t) * 16 + n16;
        qb[t] = (const uint32_t*)(qT + ((size_t)b * DIN + col) * LL);
    }

    for (int kt = 0; kt < LL / 32; ++kt) {
        HalfPack aU;
        #pragma unroll
        for (int j = 0; j < 8; ++j) {
            const int k = 32 * kt + 2 * (j & 3) + ((j & 4) ? 16 : 0) + hh * 8;
            aU.u[j] = Su[(n16 * SP + k) >> 1];   // conflict-free (stride 5 mod 64)
        }
        if (kt + 1 < LL / 32) {
            #pragma unroll
            for (int t = 0; t < 4; ++t)          // gfx1250 global_prefetch path
                __builtin_prefetch(qb[t] + ((32 * (kt + 1) + hh * 16) >> 1), 0, 1);
        }
        #pragma unroll
        for (int t = 0; t < 4; ++t) {
            HalfPack bU;
            #pragma unroll
            for (int j = 0; j < 8; ++j) {
                const int k = 32 * kt + 2 * (j & 3) + ((j & 4) ? 16 : 0) + hh * 8;  // pair in qT, table order
                bU.u[j] = qb[t][k >> 1];
            }
            acc[t] = __builtin_amdgcn_wmma_f32_16x16x32_f16(
                false, aU.v, false, bU.v, (short)0, acc[t], false, false);
        }
    }

    // ---- Epilogue: apply deferred 1/rowsum, write out [B, L, DIN] f32 ----
    float rv[8];
    #pragma unroll
    for (int r = 0; r < 8; ++r) rv[r] = rowinv[r + hh * 8];
    #pragma unroll
    for (int t = 0; t < 4; ++t) {
        const int col = (wave + 4 * t) * 16 + n16;
        #pragma unroll
        for (int r = 0; r < 8; ++r) {
            const int m = r + hh * 8;
            out[((size_t)b * LL + c0 + m) * DIN + col] = acc[t][r] * rv[r];
        }
    }
}

// ---------------------------------------------------------------------------
extern "C" void kernel_launch(void* const* d_in, const int* in_sizes, int n_in,
                              void* d_out, int out_size, void* d_ws, size_t ws_size,
                              hipStream_t stream) {
    (void)in_sizes; (void)n_in; (void)out_size; (void)ws_size;
    const float* context = (const float*)d_in[0];
    const float* query   = (const float*)d_in[1];
    const float* Win     = (const float*)d_in[2];
    const float* Wmem    = (const float*)d_in[3];
    const int*   qmask   = (const int*)d_in[4];
    float*       out     = (float*)d_out;

    _Float16* inp16 = (_Float16*)d_ws;                         // B*L*DATT halfs
    _Float16* mem16 = inp16 + (size_t)BB * LL * DATT;          // B*L*DATT halfs
    _Float16* qT    = mem16 + (size_t)BB * LL * DATT;          // B*DIN*L halfs

    proj_kernel<<<BB * LL, 128, 0, stream>>>(context, query, Win, Wmem, inp16, mem16);
    transposeq_kernel<<<BB * DIN, 256, 0, stream>>>(query, qT);

    const size_t lds_bytes = (size_t)16 * SP * 2 + 16 * sizeof(float); // ~64.4 KB
    attn_kernel<<<BB * (LL / 16), 128, lds_bytes, stream>>>(inp16, mem16, qT, qmask, out);
}
